// MPK_25589415149689
// MI455X (gfx1250) — matrix operand order repaired and checked by reading, not compiled, not run on hardware
//
#include <hip/hip_runtime.h>
#include <math.h>

// ---------------------------------------------------------------------------
// x: (2, 128, 128, 128) f32. P=8 patch-batches, C=128, HW=4096.
// Fused two-pass softmax attention on v_wmma_f32_16x16x32_bf16.
// Workspace (<20 MB, L2-resident):
//   KT  [p][n][c] bf16   8 MiB @ 0        (A of kv-GEMM, rows n)
//   Kbf [p][c][n] bf16   8 MiB @ 8 MiB    (A of out-GEMM, rows c)
//   VT  [bi][m][c] bf16  2 MiB @ 16 MiB   (B of kv-GEMM, 2x2-avg V)
//   Mrow/Srow [p][n] f32 2*128 KiB @ 18 MiB
// Block = 8 waves sharing one p; K/V tiles staged into LDS with
// GLOBAL_LOAD_ASYNC_TO_LDS_B128 (double-buffered, s_wait_asynccnt).
// ---------------------------------------------------------------------------

typedef __bf16 bf16;
typedef __attribute__((ext_vector_type(16))) __bf16 v16bf;
typedef __attribute__((ext_vector_type(8)))  __bf16 v8bf;
typedef __attribute__((ext_vector_type(8)))  float  v8f;

#define NC   128
#define NHW  4096

#if __has_builtin(__builtin_amdgcn_global_load_async_to_lds_b128) && \
    __has_builtin(__builtin_amdgcn_s_wait_asynccnt)
#define HAVE_ASYNC_LDS 1
#else
#define HAVE_ASYNC_LDS 0
#endif

#if HAVE_ASYNC_LDS
// Exact parameter types per hipcc diagnostic: v4i (vector_size 16) pointers,
// global (addrspace(1)) source first, LDS (addrspace(3)) destination second.
typedef int v4i_vs __attribute__((vector_size(16)));
typedef __attribute__((address_space(1))) v4i_vs* as1_v4i_ptr;
typedef __attribute__((address_space(3))) v4i_vs* as3_v4i_ptr;
#endif

// 16-byte global -> LDS copy (async on gfx1250).
__device__ __forceinline__ void cp16(bf16* dst_lds, const bf16* src_glob) {
#if HAVE_ASYNC_LDS
  __builtin_amdgcn_global_load_async_to_lds_b128(
      (as1_v4i_ptr)(src_glob), (as3_v4i_ptr)(dst_lds), 0, 0);
#else
  *(v8bf*)dst_lds = *(const v8bf*)src_glob;
#endif
}
__device__ __forceinline__ void cp_wait_all() {
#if HAVE_ASYNC_LDS
  __builtin_amdgcn_s_wait_asynccnt(0);
#endif
}

__device__ __forceinline__ bf16 f2bf(float f) {
  unsigned u = __float_as_uint(f);
  unsigned r = (u + 0x7FFFu + ((u >> 16) & 1u)) >> 16;   // RNE
  unsigned short s = (unsigned short)r;
  return __builtin_bit_cast(__bf16, s);
}

// A fragment (16x32 bf16, MxK): lane = hi*16 + m; two 16B chunks per lane:
//   e in [0,16) -> K = hi*16 + e
__device__ __forceinline__ v16bf load_a_frag(const bf16* base, int hi) {
  v8bf lo = *(const v8bf*)(base + hi * 16);
  v8bf hh = *(const v8bf*)(base + hi * 16 + 8);
  v16bf r;
#pragma unroll
  for (int e = 0; e < 8; ++e) { r[e] = lo[e]; r[e + 8] = hh[e]; }
  return r;
}

// ---------------------------------------------------------------------------
// Prep 1: K matrices (the loc patches), both layouts.
// ---------------------------------------------------------------------------
__global__ void MPK_prep_k(const float* __restrict__ x,
                           bf16* __restrict__ KT, bf16* __restrict__ Kbf) {
  int t = blockIdx.x * blockDim.x + threadIdx.x;   // 2^22 threads
  int n = t & 4095;
  int c = (t >> 12) & 127;
  int p = t >> 19;
  int bi = p & 1, wg = (p >> 1) & 1, hg = (p >> 2) & 1;
  int i = n >> 6, j = n & 63;
  size_t xi = (((size_t)(bi * NC + c)) * 128 + (hg << 6) + i) * 128 + (wg << 6) + j;
  bf16 v = f2bf(x[xi]);
  KT [(size_t)(p * NHW + n) * NC + c] = v;
  Kbf[(size_t)(p * NC  + c) * NHW + n] = v;
}

// ---------------------------------------------------------------------------
// Prep 2: V = bilinear 0.5x downscale == 2x2 average.  VT[bi][m][c] bf16.
// ---------------------------------------------------------------------------
__global__ void MPK_prep_v(const float* __restrict__ x, bf16* __restrict__ VT) {
  int t = blockIdx.x * blockDim.x + threadIdx.x;   // 2^20 threads
  int m = t & 4095;
  int c = (t >> 12) & 127;
  int bi = t >> 19;
  int i = m >> 6, j = m & 63;
  const float* xb = x + ((size_t)(bi * NC + c)) * 128 * 128;
  size_t r0 = (size_t)(2 * i) * 128 + 2 * j;
  float s = xb[r0] + xb[r0 + 1] + xb[r0 + 128] + xb[r0 + 129];
  VT[(size_t)(bi * NHW + m) * NC + c] = f2bf(0.25f * s);
}

// ---------------------------------------------------------------------------
// Pass 1: row stats of kv[p] = K^T V.  One wave per 16-row n-tile; all 8
// waves of a block share p, so the 16x128 V tile is staged once per block
// (async, double-buffered). Softmax state is PER-LANE (each lane owns its
// 256 columns); a single 4-step log-sum-exp shuffle merge runs at the end.
// ---------------------------------------------------------------------------
__global__ void MPK_stats(const bf16* __restrict__ KT, const bf16* __restrict__ VT,
                          float* __restrict__ Mrow, float* __restrict__ Srow) {
  __shared__ alignas(64) bf16 Vtile[2][16 * NC];   // 2 x 4 KB

  int tid  = threadIdx.x;
  int lane = tid & 31, wv = tid >> 5;
  int task = blockIdx.x * 8 + wv;          // 2048 waves total
  int p  = task >> 8;
  int n0 = (task & 255) << 4;
  int l16 = lane & 15, hi = lane >> 4;
  int bi = p & 1;

  const bf16* vbase = VT + (size_t)bi * NHW * NC;

  // Persistent A fragments (K^T rows n0..n0+15).
  const bf16* arow = KT + (size_t)(p * NHW + n0 + l16) * NC;
  v16bf A[4];
#pragma unroll
  for (int cc = 0; cc < 4; ++cc) A[cc] = load_a_frag(arow + cc * 32, hi);

  // Cooperative stage of one 16x128 V tile: 256 threads x 16 bytes.
  int srow = tid >> 4, sseg = tid & 15;
  auto stage = [&](int buf, int m0) {
    cp16(&Vtile[buf][srow * NC + sseg * 8],
         vbase + (size_t)(m0 + srow) * NC + sseg * 8);
  };

  float Ml[8], Sl[8];
#pragma unroll
  for (int r = 0; r < 8; ++r) { Ml[r] = -3.0e38f; Sl[r] = 0.f; }

  stage(0, 0);
  for (int it = 0; it < 256; ++it) {
    cp_wait_all();
    __syncthreads();                       // tile it ready; tile it-1 free
    if (it + 1 < 256) stage((it + 1) & 1, (it + 1) << 4);

    const bf16* bt = &Vtile[it & 1][l16 * NC];
    v8f d = {};
#pragma unroll
    for (int cc = 0; cc < 4; ++cc) {
      v16bf B = *(const v16bf*)(bt + cc * 32 + hi * 16);
      d = __builtin_amdgcn_wmma_f32_16x16x32_bf16(false, A[cc], false, B,
                                                  (short)0, d, false, false);
    }
    // per-lane online softmax over this lane's column
#pragma unroll
    for (int r = 0; r < 8; ++r) {
      float nm = fmaxf(Ml[r], d[r]);
      Sl[r] = Sl[r] * __expf(Ml[r] - nm) + __expf(d[r] - nm);
      Ml[r] = nm;
    }
  }

  // Merge 16 per-lane (max,sum) pairs within each hi-group (xor stays in group).
#pragma unroll
  for (int off = 1; off < 16; off <<= 1)
#pragma unroll
    for (int r = 0; r < 8; ++r) {
      float Mo = __shfl_xor(Ml[r], off, 32);
      float So = __shfl_xor(Sl[r], off, 32);
      float nm = fmaxf(Ml[r], Mo);
      Sl[r] = Sl[r] * __expf(Ml[r] - nm) + So * __expf(Mo - nm);
      Ml[r] = nm;
    }
  if (l16 == 0) {
#pragma unroll
    for (int r = 0; r < 8; ++r) {
      int n = n0 + r + 8 * hi;
      Mrow[p * NHW + n] = Ml[r];
      Srow[p * NHW + n] = Sl[r];
    }
  }
}

// ---------------------------------------------------------------------------
// Pass 2: out[p][c][m] = sum_n K[c][n] * exp(kv[n][m]-M[n])/S[n]; residuals
// fused (d_out pixel = attn_out + 2*x). One wave per 16-wide m-tile; the 8
// waves of a block share p, so the per-n0 K tiles (KT rows for WMMA1, Kbf
// columns for WMMA2) are staged once per block, async + double-buffered.
// ---------------------------------------------------------------------------
__global__ void MPK_out(const float* __restrict__ x,
                        const bf16* __restrict__ KT, const bf16* __restrict__ Kbf,
                        const bf16* __restrict__ VT,
                        const float* __restrict__ Mrow, const float* __restrict__ Srow,
                        float* __restrict__ out) {
  __shared__ alignas(64) bf16 KTt[2][32 * NC];   // 2 x 8 KB: [n_local][c]
  __shared__ alignas(64) bf16 KBt[2][NC * 32];   // 2 x 8 KB: [c][n_local]
  __shared__ alignas(64) bf16 Pt[8][16 * 32];    // per-wave P tile [m_l][n_l]

  int tid  = threadIdx.x;
  int lane = tid & 31, wv = tid >> 5;
  int task = blockIdx.x * 8 + wv;          // 2048 waves total
  int p  = task >> 8;
  int m0 = (task & 255) << 4;
  int l16 = lane & 15, hi = lane >> 4;
  int bi = p & 1, wg = (p >> 1) & 1, hg = (p >> 2) & 1;

  bf16* pt = &Pt[wv][0];

  // Cooperative stage: KT rows [n0,n0+32) and Kbf columns [n0,n0+32).
  int tr = tid >> 3, tch = tid & 7;        // KTt: 32 rows x 8 chunks of 16
  int tc = tid >> 1, thf = tid & 1;        // KBt: 128 rows x 2 halves of 16
  auto stage = [&](int buf, int n0) {
    const bf16* s0 = KT + (size_t)(p * NHW + n0 + tr) * NC + tch * 16;
    bf16* d0 = &KTt[buf][tr * NC + tch * 16];
    cp16(d0, s0); cp16(d0 + 8, s0 + 8);
    const bf16* s1 = Kbf + (size_t)(p * NC + tc) * NHW + n0 + thf * 16;
    bf16* d1 = &KBt[buf][tc * 32 + thf * 16];
    cp16(d1, s1); cp16(d1 + 8, s1 + 8);
  };

  // Loop-invariant B1 fragments (V^T rows of this m-tile) straight from L2.
  const bf16* brow = VT + (size_t)(bi * NHW + m0 + l16) * NC;
  v16bf B1[4];
#pragma unroll
  for (int cc = 0; cc < 4; ++cc) B1[cc] = *(const v16bf*)(brow + cc * 32 + hi * 16);

  v8f acc[8];
#pragma unroll
  for (int ct = 0; ct < 8; ++ct) acc[ct] = (v8f){};

  const float* Mp = Mrow + p * NHW;
  const float* Sp = Srow + p * NHW;

  stage(0, 0);
  for (int it = 0; it < 128; ++it) {
    int n0  = it << 5;
    int buf = it & 1;
    cp_wait_all();
    __syncthreads();                       // tile it ready; tile it-1 free
    if (it + 1 < 128) stage(buf ^ 1, n0 + 32);

    // --- WMMA1: 32x16 kv sub-tile from LDS, exp/scale into per-wave Pt ---
#pragma unroll
    for (int sub = 0; sub < 2; ++sub) {
      const bf16* ar = &KTt[buf][(sub * 16 + l16) * NC];
      v8f d = {};
#pragma unroll
      for (int cc = 0; cc < 4; ++cc) {
        v16bf A1 = load_a_frag(ar + cc * 32, hi);
        d = __builtin_amdgcn_wmma_f32_16x16x32_bf16(false, A1, false, B1[cc],
                                                    (short)0, d, false, false);
      }
#pragma unroll
      for (int r = 0; r < 8; ++r) {
        int nl = sub * 16 + r + 8 * hi;
        int n  = n0 + nl;
        float pv = __expf(d[r] - Mp[n]) * (1.0f / Sp[n]);
        pt[l16 * 32 + nl] = f2bf(pv);      // transposed: Pt[m_local][n_local]
      }
    }
    __builtin_amdgcn_wave_barrier();       // DS in-order per wave; pin schedule

    v16bf B2 = *(const v16bf*)(pt + l16 * 32 + hi * 16);

    // --- WMMA2: accumulate 128x16 output tile from LDS-resident Kbf ---
#pragma unroll
    for (int ct = 0; ct < 8; ++ct) {
      v16bf A2 = load_a_frag(&KBt[buf][(ct * 16 + l16) * 32], hi);
      acc[ct] = __builtin_amdgcn_wmma_f32_16x16x32_bf16(false, A2, false, B2,
                                                        (short)0, acc[ct], false, false);
    }
    __builtin_amdgcn_wave_barrier();
  }

  // Epilogue: patches2image placement + both residuals (out + loc + x = out+2x).
  int m  = m0 + l16;
  int gi = (hg << 6) + (m >> 6);
  int gj = (wg << 6) + (m & 63);
#pragma unroll
  for (int ct = 0; ct < 8; ++ct)
#pragma unroll
    for (int r = 0; r < 8; ++r) {
      int c = ct * 16 + r + 8 * hi;
      size_t idx = (((size_t)(bi * NC + c)) * 128 + gi) * 128 + gj;
      out[idx] = acc[ct][r] + 2.0f * x[idx];
    }
}

// ---------------------------------------------------------------------------
extern "C" void kernel_launch(void* const* d_in, const int* in_sizes, int n_in,
                              void* d_out, int out_size, void* d_ws, size_t ws_size,
                              hipStream_t stream) {
  const float* x = (const float*)d_in[0];
  float* out = (float*)d_out;

  char* ws = (char*)d_ws;
  bf16*  KT   = (bf16*)(ws);                                // 8 MiB
  bf16*  Kbf  = (bf16*)(ws + ((size_t)8  << 20));           // 8 MiB
  bf16*  VT   = (bf16*)(ws + ((size_t)16 << 20));           // 2 MiB
  float* Mrow = (float*)(ws + ((size_t)18 << 20));          // 128 KiB
  float* Srow = (float*)(ws + ((size_t)18 << 20) + 131072); // 128 KiB

  MPK_prep_k<<<16384, 256, 0, stream>>>(x, KT, Kbf);
  MPK_prep_v<<<4096,  256, 0, stream>>>(x, VT);
  MPK_stats <<<256,   256, 0, stream>>>(KT, VT, Mrow, Srow);
  MPK_out   <<<256,   256, 0, stream>>>(x, KT, Kbf, VT, Mrow, Srow, out);
}
